// WaveNet_70025146794367
// MI455X (gfx1250) — matrix operand order repaired and checked by reading, not compiled, not run on hardware
//
#include <hip/hip_runtime.h>
#include <hip/hip_bf16.h>

// Fast-WaveNet restructuring for MI455X (gfx1250, wave32, WMMA).
//  - Encoder TCN pass once (bulk of FLOPs) as bf16 WMMA GEMMs.
//  - Incremental one-column-per-layer updates for the 32 autoregressive steps.
//  - Activations stored [t][B][C] bf16 so WMMA B-operand loads are contiguous.
//  - Conv weights pre-packed into 16x32 bf16 A tiles, k contiguous per lane half.
//  - All WMMA kernels use 1-wave (32-thread) blocks => all control flow is
//    scalar, EXEC stays all-ones as the ISA requires.

typedef __bf16 bf16_t;
typedef __attribute__((ext_vector_type(16))) __bf16 v16bf;
typedef __attribute__((ext_vector_type(8)))  __bf16 v8bf;
typedef __attribute__((ext_vector_type(8)))  float  v8f;

#define BATCH    64
#define CCH      256
#define HID      512
#define ENC      512
#define TBUF     544   // ENC + DEC_LEN
#define INLEN    543   // ENC + DEC_LEN - 1

// ---------------------------------------------------------------------------
// Weight packing: convs_w [7][256][256][2] f32 -> bf16 A-layout tiles
// PW[j][tap][mt][kt][lane][e], 16x32 tile (M=co, K=ci chunk of 32).
// A-layout (16-bit A 16x32): lane half selects K 0..15 / 16..31, contiguous per lane
// (matches the contiguous B-operand loads).
// ---------------------------------------------------------------------------
__global__ void pack_conv_kernel(const float* __restrict__ W, bf16_t* __restrict__ PW) {
    int idx = blockIdx.x * 256 + threadIdx.x;       // 7*2*16*8*32*16 = 917504
    if (idx >= 917504) return;
    int e    = idx & 15;
    int lane = (idx >> 4) & 31;
    int kt   = (idx >> 9) & 7;
    int mt   = (idx >> 12) & 15;
    int tap  = (idx >> 16) & 1;
    int j    = idx >> 17;
    int co   = mt * 16 + (lane & 15);
    int ci   = kt * 32 + (lane >> 4) * 16 + e;
    float v  = W[((size_t)(j * CCH + co) * CCH + ci) * 2 + tap];
    PW[idx] = (bf16_t)v;
}

// w1 [256][512] f32 -> bf16 B-layout tiles PW1[ht][kt][lane][e]
// B-layout (32x16): lanes 0-15 hold K {0..7,16..23}, lanes 16-31 K {8..15,24..31}.
__global__ void pack_w1_kernel(const float* __restrict__ W1, bf16_t* __restrict__ PW1) {
    int idx = blockIdx.x * 256 + threadIdx.x;       // 32*8*32*16 = 131072
    if (idx >= 131072) return;
    int e    = idx & 15;
    int lane = (idx >> 4) & 31;
    int kt   = (idx >> 9) & 7;
    int ht   = idx >> 12;
    int c    = kt * 32 + ((e < 8) ? e : e + 8) + ((lane >> 4) ? 8 : 0);
    int h    = ht * 16 + (lane & 15);
    PW1[idx] = (bf16_t)W1[(size_t)c * HID + h];
}

// buf[b][t] = inputs[b][t] for t<512, else 0 (future feedback slots)
__global__ void init_buf_kernel(const float* __restrict__ in, float* __restrict__ buf) {
    int idx = blockIdx.x * 256 + threadIdx.x;
    if (idx >= BATCH * TBUF) return;
    int b = idx / TBUF, t = idx % TBUF;
    buf[idx] = (t < ENC) ? in[(size_t)b * INLEN + t] : 0.0f;
}

// Layer 0: 1 -> 256 channels, K=2, dilation 1 (tiny, pure VALU).
// Writes bf16 activations X0[t][b][c].
__global__ void layer0_kernel(const float* __restrict__ buf,
                              const float* __restrict__ w0,   // [256][1][2]
                              const float* __restrict__ b0,   // [256]
                              bf16_t* __restrict__ X0, int t0) {
    int co  = threadIdx.x;                 // 256
    int blk = blockIdx.x;                  // nT*64
    int b   = blk & 63;
    int t   = t0 + (blk >> 6);
    float xm1 = (t >= 1) ? buf[(size_t)b * TBUF + t - 1] : 0.0f;
    float xc  = buf[(size_t)b * TBUF + t];
    float y = w0[co * 2 + 0] * xm1 + w0[co * 2 + 1] * xc + b0[co];
    X0[(size_t)(t * BATCH + b) * CCH + co] = (bf16_t)y;
}

// ---------------------------------------------------------------------------
// Dilated causal conv layer as WMMA GEMM.
//   y[co, (b,t)] = sum_ci W0[co,ci]*x[ci,(b,t-d)] + W1[co,ci]*x[ci,(b,t)] + bias
// One wave per block; each wave: 16 co x 16 b tile, 4 consecutive t (A reuse x4).
// ---------------------------------------------------------------------------
__global__ __launch_bounds__(32) void conv_wmma_kernel(
    const bf16_t* __restrict__ Xin,   // [t][64][256] bf16
    const bf16_t* __restrict__ PW,    // packed A tiles [tap][mt][kt][lane][16]
    const float*  __restrict__ bias,  // [256]
    bf16_t* __restrict__ Xout,        // [t][64][256] bf16
    int dil, int t0, int nT)
{
    const int lane = threadIdx.x;
    const int wid  = blockIdx.x;
    const int mt = wid & 15;          // 16 co-tiles
    const int bt = (wid >> 4) & 3;    // 4 batch-tiles
    const int tg = wid >> 6;          // t-groups of 4
    const int tbase = t0 + tg * 4;
    int ntt = nT - tg * 4; if (ntt > 4) ntt = 4;

    const int b     = bt * 16 + (lane & 15);   // B-operand column
    const int khalf = (lane >> 4) * 16;        // B-operand K half

    v8f acc[4] = {};

    #pragma unroll
    for (int tap = 0; tap < 2; ++tap) {
        #pragma unroll
        for (int kt = 0; kt < 8; ++kt) {
            const v16bf a = *(const v16bf*)(PW + ((size_t)((tap * 16 + mt) * 8 + kt) * 32 + lane) * 16);
            #pragma unroll
            for (int tt = 0; tt < 4; ++tt) {
                if (tt < ntt) {
                    const int t  = tbase + tt;
                    const int ts = (tap == 0) ? (t - dil) : t;   // left zero-pad
                    if (ts >= 0) {   // scalar branch: t, dil are block-uniform
                        const v16bf bm = *(const v16bf*)(Xin + (size_t)(ts * BATCH + b) * CCH + kt * 32 + khalf);
                        acc[tt] = __builtin_amdgcn_wmma_f32_16x16x32_bf16(
                            false, a, false, bm, (short)0, acc[tt], false, false);
                    }
                }
            }
        }
    }

    const int corow = mt * 16 + (lane >> 4) * 8;  // D rows held by this lane
    #pragma unroll
    for (int tt = 0; tt < 4; ++tt) {
        if (tt < ntt) {
            const int t = tbase + tt;
            v8bf o;
            #pragma unroll
            for (int v = 0; v < 8; ++v)
                o[v] = (bf16_t)(acc[tt][v] + bias[corow + v]);
            *(v8bf*)(Xout + (size_t)(t * BATCH + b) * CCH + corow) = o;  // 16B store
        }
    }
}

// MLP layer 1: hidden[b][h] = feat[b][:] @ w1 + b1 ; feat = X7 at timestep t.
// A = feat (16b x 32c, A-layout gather = two contiguous 16B runs per lane).
__global__ __launch_bounds__(32) void mlp1_wmma_kernel(
    const bf16_t* __restrict__ X7, const bf16_t* __restrict__ PW1,
    const float* __restrict__ b1, float* __restrict__ hidden, int t)
{
    const int lane = threadIdx.x;
    const int bt = blockIdx.x & 3;     // 4 batch tiles
    const int ht = blockIdx.x >> 2;    // 32 hidden tiles

    const int brow = bt * 16 + (lane & 15);
    const int aoff = (lane >> 4) ? 8 : 0;
    const bf16_t* feat = X7 + (size_t)(t * BATCH + brow) * CCH;

    v8f acc = {};
    #pragma unroll
    for (int kt = 0; kt < 8; ++kt) {
        const v8bf r0 = *(const v8bf*)(feat + kt * 32 + aoff);       // K e..e+7
        const v8bf r1 = *(const v8bf*)(feat + kt * 32 + aoff + 16);  // K +16
        v16bf a;
        #pragma unroll
        for (int v = 0; v < 8; ++v) { a[v] = r0[v]; a[v + 8] = r1[v]; }
        const v16bf bm = *(const v16bf*)(PW1 + ((size_t)(ht * 8 + kt) * 32 + lane) * 16);
        acc = __builtin_amdgcn_wmma_f32_16x16x32_bf16(
            false, a, false, bm, (short)0, acc, false, false);
    }

    const int h  = ht * 16 + (lane & 15);
    const float bv = b1[h];
    const int b0r = bt * 16 + (lane >> 4) * 8;
    #pragma unroll
    for (int v = 0; v < 8; ++v)
        hidden[(size_t)(b0r + v) * HID + h] = acc[v] + bv;
}

// MLP layer 2 + feedback: o = relu(hidden) @ w2 + b2; write d_out and buf[t+1].
__global__ void mlp2_kernel(const float* __restrict__ hidden,
                            const float* __restrict__ w2, const float* __restrict__ b2,
                            float* __restrict__ buf, float* __restrict__ out,
                            int t, int step)
{
    __shared__ float red[256];
    const int b = blockIdx.x, tid = threadIdx.x;
    float s = 0.0f;
    for (int h = tid; h < HID; h += 256) {
        float v = hidden[(size_t)b * HID + h];
        s += fmaxf(v, 0.0f) * w2[h];
    }
    red[tid] = s; __syncthreads();
    for (int off = 128; off > 0; off >>= 1) {
        if (tid < off) red[tid] += red[tid + off];
        __syncthreads();
    }
    if (tid == 0) {
        float o = red[0] + b2[0];
        out[(size_t)b * 32 + step] = o;
        buf[(size_t)b * TBUF + (t + 1)] = o;   // autoregressive feedback
    }
}

extern "C" void kernel_launch(void* const* d_in, const int* in_sizes, int n_in,
                              void* d_out, int out_size, void* d_ws, size_t ws_size,
                              hipStream_t stream) {
    const float* inp     = (const float*)d_in[0];
    const float* conv0_w = (const float*)d_in[1];
    const float* conv0_b = (const float*)d_in[2];
    const float* convs_w = (const float*)d_in[3];
    const float* convs_b = (const float*)d_in[4];
    const float* w1      = (const float*)d_in[5];
    const float* b1      = (const float*)d_in[6];
    const float* w2      = (const float*)d_in[7];
    const float* b2      = (const float*)d_in[8];
    float* out = (float*)d_out;

    // Workspace carve-up (~145 MB)
    char* ws = (char*)d_ws;
    size_t off = 0;
    auto carve = [&](size_t bytes) -> void* {
        void* p = ws + off; off += (bytes + 255) & ~(size_t)255; return p;
    };
    float*  buf    = (float*)carve((size_t)BATCH * TBUF * 4);
    float*  hidden = (float*)carve((size_t)BATCH * HID * 4);
    bf16_t* PW     = (bf16_t*)carve((size_t)917504 * 2);
    bf16_t* PW1    = (bf16_t*)carve((size_t)131072 * 2);
    bf16_t* X[8];
    for (int j = 0; j < 8; ++j)
        X[j] = (bf16_t*)carve((size_t)TBUF * BATCH * CCH * 2);

    static const int DIL[8] = {1, 2, 4, 8, 16, 32, 64, 128};

    // One-shot packs + buffer init
    pack_conv_kernel<<<917504 / 256, 256, 0, stream>>>(convs_w, PW);
    pack_w1_kernel<<<131072 / 256, 256, 0, stream>>>(w1, PW1);
    init_buf_kernel<<<(BATCH * TBUF + 255) / 256, 256, 0, stream>>>(inp, buf);

    // Encoder pass over t in [0, 512)
    layer0_kernel<<<ENC * BATCH, 256, 0, stream>>>(buf, conv0_w, conv0_b, X[0], 0);
    for (int j = 1; j < 8; ++j)
        conv_wmma_kernel<<<16 * 4 * (ENC / 4), 32, 0, stream>>>(
            X[j - 1], PW + (size_t)(j - 1) * 131072, convs_b + (j - 1) * CCH,
            X[j], DIL[j], 0, ENC);

    // Autoregressive decode: 32 steps, incremental one-column TCN updates
    for (int i = 0; i < 32; ++i) {
        const int t = ENC - 1 + i;
        mlp1_wmma_kernel<<<128, 32, 0, stream>>>(X[7], PW1, b1, hidden, t);
        mlp2_kernel<<<BATCH, 256, 0, stream>>>(hidden, w2, b2, buf, out, t, i);
        if (i < 31) {
            const int tn = t + 1;
            layer0_kernel<<<BATCH, 256, 0, stream>>>(buf, conv0_w, conv0_b, X[0], tn);
            for (int j = 1; j < 8; ++j)
                conv_wmma_kernel<<<16 * 4, 32, 0, stream>>>(
                    X[j - 1], PW + (size_t)(j - 1) * 131072, convs_b + (j - 1) * CCH,
                    X[j], DIL[j], tn, 1);
        }
    }
}
